// GCN_76802605187214
// MI455X (gfx1250) — matrix operand order repaired and checked by reading, not compiled, not run on hardware
//
#include <hip/hip_runtime.h>
#include <hip/hip_bf16.h>

typedef __attribute__((ext_vector_type(16))) _Float16 v16h;
typedef __attribute__((ext_vector_type(8)))  float    v8f;
typedef __attribute__((ext_vector_type(4)))  float    v4f;

#define D 64

// ---------------------------------------------------------------- utilities
__global__ void fill_zero_kernel(float* __restrict__ p, size_t n) {
    size_t i = (size_t)blockIdx.x * blockDim.x + threadIdx.x;
    if (i < n) p[i] = 0.0f;
}

__global__ void fill_zero4_kernel(v4f* __restrict__ p, size_t n4) {
    size_t i = (size_t)blockIdx.x * blockDim.x + threadIdx.x;
    if (i < n4) p[i] = (v4f){0.0f, 0.0f, 0.0f, 0.0f};
}

// deg[dst] += 1 per edge (deg buffer pre-zeroed)
__global__ void degree_kernel(const int* __restrict__ dst, float* __restrict__ deg, int E) {
    int e = blockIdx.x * blockDim.x + threadIdx.x;
    if (e < E) atomicAdd(&deg[dst[e]], 1.0f);
}

// dinv[i] = rsqrt(deg[i] + 1)   (self-loop)
__global__ void rsqrt_kernel(float* __restrict__ d, int n) {
    int i = blockIdx.x * blockDim.x + threadIdx.x;
    if (i < n) d[i] = rsqrtf(d[i] + 1.0f);
}

// ------------------------------------------------- W -> f16 B-fragment pack
// Output layout: [kchunk c=0..1][ntile=0..3][lane=0..31][half h=0..15],
// so a wave's B fragment is one contiguous 32-byte v16h per lane.
// Element: Wf[(c,nt,lane,h)] = W[(c*32 + (lane>>4)*16 + h)*64 + nt*16 + (lane&15)]
__global__ void pack_w_f16_kernel(const float* __restrict__ W, _Float16* __restrict__ Wf) {
    const int tid  = threadIdx.x;        // 256 threads, one per (c,nt,lane)
    const int c    = tid >> 7;           // 0..1
    const int nt   = (tid >> 5) & 3;     // 0..3
    const int lane = tid & 31;           // 0..31
    const int ks   = c * 32 + (lane >> 4) * 16;
    const int n    = nt * 16 + (lane & 15);
    _Float16* out  = Wf + (size_t)tid * 16;
    #pragma unroll
    for (int h = 0; h < 16; ++h)
        out[h] = (_Float16)W[(size_t)(ks + h) * D + n];
}

// ------------------------------------------------------ WMMA GEMM: Y = X @ W
// X: [nrows, 64] f32 row-major, Wf: pre-packed f16 B fragments, Y: [nrows,64].
// Block = 256 threads = 8 waves; wave (mtile 0..1, ntile 0..3) owns a 16x16
// tile; K = 64 -> two v_wmma_f32_16x16x32_f16.
__global__ __launch_bounds__(256) void gemm64_wmma_kernel(
    const float* __restrict__ X, const _Float16* __restrict__ Wf,
    float* __restrict__ Y, int nrows)
{
    const int lane    = threadIdx.x & 31;
    const int wave    = threadIdx.x >> 5;
    const int mtile   = wave >> 2;         // 0..1
    const int ntile   = wave & 3;          // 0..3
    const int rowBase = blockIdx.x * 32 + mtile * 16;
    const int colBase = ntile * 16;
    const int halfSel = lane >> 4;         // 0: lanes 0-15, 1: lanes 16-31
    const int l15     = lane & 15;

    // clamp loads so EXEC stays all-ones through the WMMAs
    int mrow = rowBase + l15;
    if (mrow >= nrows) mrow = nrows - 1;
    const v4f* xr4 = (const v4f*)(X + (size_t)mrow * D);
    const v16h* wp = (const v16h*)Wf;

    v8f acc = {};
    #pragma unroll
    for (int k0 = 0; k0 < D; k0 += 32) {
        const int c  = k0 >> 5;
        const int kb = k0 + halfSel * 16;  // lanes 16-31 start at K=16
        // ---- A fragment (16x32 f16, contiguous k): halves 0-7 <- K kb..kb+7,
        //      halves 8-15 <- K kb+8..kb+15; one 64B (4xfloat4) segment ----
        v4f p0 = xr4[(kb >> 2) + 0];
        v4f p1 = xr4[(kb >> 2) + 1];
        v4f q0 = xr4[((kb + 8) >> 2) + 0];
        v4f q1 = xr4[((kb + 8) >> 2) + 1];
        v16h a;
        #pragma unroll
        for (int i = 0; i < 4; ++i) {
            a[i]      = (_Float16)p0[i];
            a[4 + i]  = (_Float16)p1[i];
            a[8 + i]  = (_Float16)q0[i];
            a[12 + i] = (_Float16)q1[i];
        }
        // ---- B fragment: one contiguous 32B load from the packed buffer ----
        v16h b = wp[(c * 4 + ntile) * 32 + lane];

        acc = __builtin_amdgcn_wmma_f32_16x16x32_f16(
            /*neg_a=*/false, a, /*neg_b=*/false, b,
            /*c_mod=*/(short)0, acc, /*reuse_a=*/false, /*reuse_b=*/false);
    }

    // ---- store C/D: VGPR r -> row rowBase + halfSel*8 + r, col colBase+l15
    const int n     = colBase + l15;
    const int mBase = rowBase + halfSel * 8;
    if (rowBase + 16 <= nrows) {           // wave-uniform fast path
        #pragma unroll
        for (int r = 0; r < 8; ++r)
            Y[(size_t)(mBase + r) * D + n] = acc[r];
    } else {
        #pragma unroll
        for (int r = 0; r < 8; ++r) {
            const int row = mBase + r;
            if (row < nrows) Y[(size_t)row * D + n] = acc[r];
        }
    }
}

// ------------------------------------------- edge scatter: agg[dst] += c*xw[src]
// 32 lanes per edge, two features per lane (float2 load, 2 f32 atomics).
// Edge id is wave-uniform -> scalar src/dst/dinv loads. agg pre-zeroed.
__global__ __launch_bounds__(256) void scatter_agg_kernel(
    const int* __restrict__ src, const int* __restrict__ dst,
    const float* __restrict__ dinv, const float* __restrict__ xw,
    float* __restrict__ agg, int E)
{
    long long t = (long long)blockIdx.x * blockDim.x + threadIdx.x;
    int e = (int)(t >> 5);
    if (e >= E) return;
    int j2 = (int)(t & 31) * 2;
    int s = src[e], d = dst[e];
    float c = dinv[s] * dinv[d];
    float2 v = *(const float2*)(xw + (size_t)s * D + j2);
    float* ap = agg + (size_t)d * D + j2;
    atomicAdd(ap,     c * v.x);
    atomicAdd(ap + 1, c * v.y);
}

// ------------------- fused: h = agg + dinv^2 * xw + bias ; column sum / sumsq
// thread -> 4 columns (float4), rows strided; 8 atomicAdds per thread at end.
__global__ __launch_bounds__(256) void selfloop_bias_stats_kernel(
    float* __restrict__ h /* in: agg, out: h (in place) */,
    const float* __restrict__ xw, const float* __restrict__ dinv,
    const float* __restrict__ bias,
    float* __restrict__ sums, float* __restrict__ sumsq, int nrows)
{
    const int j4       = (threadIdx.x & 15) * 4;
    const int rowInBlk = threadIdx.x >> 4;           // 0..15
    const int rowsStep = gridDim.x * 16;
    const v4f bj = *(const v4f*)(bias + j4);
    v4f s = {0.0f, 0.0f, 0.0f, 0.0f};
    v4f q = {0.0f, 0.0f, 0.0f, 0.0f};
    for (int i = blockIdx.x * 16 + rowInBlk; i < nrows; i += rowsStep) {
        const float di = dinv[i];
        const float dd = di * di;
        const size_t idx = (size_t)i * D + j4;
        v4f a = *(const v4f*)(h + idx);
        v4f w = *(const v4f*)(xw + idx);
        v4f v = a + dd * w + bj;
        *(v4f*)(h + idx) = v;
        s += v;
        q += v * v;
    }
    #pragma unroll
    for (int k = 0; k < 4; ++k) {
        atomicAdd(&sums[j4 + k],  s[k]);
        atomicAdd(&sumsq[j4 + k], q[k]);
    }
}

// scale[j] = gamma*rsqrt(var+eps) ; shift[j] = beta - mu*scale
__global__ void bn_finalize_kernel(
    const float* __restrict__ sums, const float* __restrict__ sumsq,
    const float* __restrict__ gamma, const float* __restrict__ beta,
    float* __restrict__ scale, float* __restrict__ shift, float invN)
{
    int j = threadIdx.x;   // 64 threads
    float mu  = sums[j] * invN;
    float var = sumsq[j] * invN - mu * mu;
    float sc  = gamma[j] * rsqrtf(var + 1e-5f);
    scale[j] = sc;
    shift[j] = beta[j] - mu * sc;
}

// out = leaky_relu(h*scale + shift, 0.01), float4 per thread
__global__ __launch_bounds__(256) void bn_lrelu_kernel(
    const v4f* __restrict__ h, const float* __restrict__ scale,
    const float* __restrict__ shift, v4f* __restrict__ out, size_t total4)
{
    size_t t = (size_t)blockIdx.x * blockDim.x + threadIdx.x;
    if (t >= total4) return;
    const int j4 = (int)(t & 15) * 4;
    v4f sc = *(const v4f*)(scale + j4);
    v4f sh = *(const v4f*)(shift + j4);
    v4f v = h[t];
    v = v * sc + sh;
    #pragma unroll
    for (int k = 0; k < 4; ++k)
        v[k] = v[k] > 0.0f ? v[k] : 0.01f * v[k];
    out[t] = v;
}

// ---------------------------------------------------------------- launcher
extern "C" void kernel_launch(void* const* d_in, const int* in_sizes, int n_in,
                              void* d_out, int out_size, void* d_ws, size_t ws_size,
                              hipStream_t stream) {
    const float* x      = (const float*)d_in[0];
    const int*   ei     = (const int*)  d_in[1];
    const float* W1     = (const float*)d_in[2];
    const float* b1     = (const float*)d_in[3];
    const float* gamma1 = (const float*)d_in[4];
    const float* beta1  = (const float*)d_in[5];
    const float* W2     = (const float*)d_in[6];
    const float* b2     = (const float*)d_in[7];
    const float* gamma2 = (const float*)d_in[8];
    const float* beta2  = (const float*)d_in[9];

    const int N = in_sizes[0] / D;
    const int E = in_sizes[1] / 2;
    const int* src = ei;
    const int* dst = ei + E;

    // workspace layout (floats)
    float*     xw    = (float*)d_ws;             // N*64
    float*     agg   = xw  + (size_t)N * D;      // N*64
    float*     dinv  = agg + (size_t)N * D;      // N
    float*     sums  = dinv + N;                 // 64
    float*     sumsq = sums + 64;                // 64
    float*     scale = sumsq + 64;               // 64
    float*     shift = scale + 64;               // 64
    _Float16*  wpack = (_Float16*)(shift + 64);  // 4096 halves (8 KB), 32B-aligned

    const size_t ND  = (size_t)N * D;
    const size_t ND4 = ND / 4;                   // D=64 -> always divisible
    const int gFill1  = (int)((N + 255) / 256);
    const int gFill4  = (int)((ND4 + 255) / 256);
    const int gEdge   = (int)(((long long)E * 32 + 255) / 256);
    const int gGemm   = (N + 31) / 32;
    const float invN  = 1.0f / (float)N;

    // ---- degrees / normalization coefficients ----
    fill_zero_kernel<<<gFill1, 256, 0, stream>>>(dinv, (size_t)N);
    degree_kernel<<<(E + 255) / 256, 256, 0, stream>>>(dst, dinv, E);
    rsqrt_kernel<<<gFill1, 256, 0, stream>>>(dinv, N);

    // =============== layer 1 ===============
    pack_w_f16_kernel<<<1, 256, 0, stream>>>(W1, wpack);
    gemm64_wmma_kernel<<<gGemm, 256, 0, stream>>>(x, wpack, xw, N);
    fill_zero4_kernel<<<gFill4, 256, 0, stream>>>((v4f*)agg, ND4);
    scatter_agg_kernel<<<gEdge, 256, 0, stream>>>(src, dst, dinv, xw, agg, E);
    fill_zero_kernel<<<1, 128, 0, stream>>>(sums, (size_t)128);  // sums+sumsq
    selfloop_bias_stats_kernel<<<864, 256, 0, stream>>>(agg, xw, dinv, b1, sums, sumsq, N);
    bn_finalize_kernel<<<1, 64, 0, stream>>>(sums, sumsq, gamma1, beta1, scale, shift, invN);
    bn_lrelu_kernel<<<gFill4, 256, 0, stream>>>((const v4f*)agg, scale, shift, (v4f*)agg, ND4);

    // =============== layer 2 ===============
    pack_w_f16_kernel<<<1, 256, 0, stream>>>(W2, wpack);
    gemm64_wmma_kernel<<<gGemm, 256, 0, stream>>>(agg, wpack, xw, N);        // xw2
    fill_zero4_kernel<<<gFill4, 256, 0, stream>>>((v4f*)agg, ND4);           // reuse as agg2
    scatter_agg_kernel<<<gEdge, 256, 0, stream>>>(src, dst, dinv, xw, agg, E);
    fill_zero_kernel<<<1, 128, 0, stream>>>(sums, (size_t)128);
    selfloop_bias_stats_kernel<<<864, 256, 0, stream>>>(agg, xw, dinv, b2, sums, sumsq, N);
    bn_finalize_kernel<<<1, 64, 0, stream>>>(sums, sumsq, gamma2, beta2, scale, shift, invN);
    bn_lrelu_kernel<<<gFill4, 256, 0, stream>>>((const v4f*)agg, scale, shift, (v4f*)d_out, ND4);
}
